// SupervisedContrastiveLoss_9285719294132
// MI455X (gfx1250) — matrix operand order repaired and checked by reading, not compiled, not run on hardware
//
#include <hip/hip_runtime.h>

typedef __attribute__((ext_vector_type(16))) _Float16 v16h;
typedef __attribute__((ext_vector_type(8)))  _Float16 v8h;
typedef __attribute__((ext_vector_type(4)))  _Float16 v4h;
typedef __attribute__((ext_vector_type(8)))  float    v8f;

#define N_ROWS 8192
#define DIM    128
#define INV_T  (1.0f / 0.07f)
#define SLICES 4
#define COLS_PER_SLICE (N_ROWS / SLICES)
#define TILES_PER_SLICE (COLS_PER_SLICE / 16)

// ---------------------------------------------------------------------------
// Kernel 1: L2-normalize rows (f32) and emit f16 copy for WMMA.
// One wave32 per row; each lane owns 4 consecutive floats (float4 load).
// ---------------------------------------------------------------------------
__global__ void __launch_bounds__(256)
scl_normalize(const float* __restrict__ feat, _Float16* __restrict__ out)
{
    const int wave = (blockIdx.x * blockDim.x + threadIdx.x) >> 5;   // row id
    const int lane = threadIdx.x & 31;
    if (wave >= N_ROWS) return;

    const float4 v = ((const float4*)(feat + (size_t)wave * DIM))[lane];
    float ss = v.x * v.x + v.y * v.y + v.z * v.z + v.w * v.w;
    #pragma unroll
    for (int m = 16; m >= 1; m >>= 1) ss += __shfl_xor(ss, m, 32);

    const float s = 1.0f / fmaxf(sqrtf(ss), 1e-12f);   // F.normalize eps

    v4h o;
    o[0] = (_Float16)(v.x * s);
    o[1] = (_Float16)(v.y * s);
    o[2] = (_Float16)(v.z * s);
    o[3] = (_Float16)(v.w * s);
    *(v4h*)(out + (size_t)wave * DIM + lane * 4) = o;
}

// B tile for one 16-column block: full K=128 per lane-half.
struct BTile { v16h b[4]; };

// ---------------------------------------------------------------------------
// Kernel 2: fused feats·featsᵀ (f16 WMMA, f32 accum) + masked softmax stats.
// Wave owns 16 rows; block = 8 waves = 128 rows. blockIdx.y = column slice.
// A (16x128) is register-resident; B tiles are double-buffered so the next
// tile's 8x global_load_b128 are in flight during the WMMAs + exp epilogue.
// Per element: x = (dot-1)/T (constant max-shift since rowmax = l_ii = 1/T),
//   Z += exp(x) [j!=i];  S += x, P += 1 [label match, j!=i].
// ---------------------------------------------------------------------------
__global__ void __launch_bounds__(256)
scl_main(const _Float16* __restrict__ fh, const int* __restrict__ labels,
         float* __restrict__ pZ, float* __restrict__ pS, float* __restrict__ pP)
{
    const int lane   = threadIdx.x & 31;
    const int wave   = threadIdx.x >> 5;
    const int lid    = lane & 15;      // column-within-tile / row-within-tile
    const int hi     = lane >> 4;      // half-wave select
    const int base_i = (blockIdx.x * 8 + wave) * 16;
    const int col0   = blockIdx.y * COLS_PER_SLICE;

    // --- A tile (16 rows x K=128), WMMA 16-bit A layout ---
    // lanes 0-15 : row M=lid, K = kc*32 + {0..7, 16..23}
    // lanes 16-31: row M=lid, K = kc*32 + {8..15, 24..31}
    const _Float16* arow = fh + (size_t)(base_i + lid) * DIM;
    v16h A[4];
    #pragma unroll
    for (int kc = 0; kc < 4; ++kc) {
        const v8h lo  = *(const v8h*)(arow + kc * 32 + hi * 8);
        const v8h hi8 = *(const v8h*)(arow + kc * 32 + 16 + hi * 8);
        #pragma unroll
        for (int k = 0; k < 8; ++k) { A[kc][k] = lo[k]; A[kc][k + 8] = hi8[k]; }
    }

    // labels of the rows this lane's C registers correspond to
    int labI[8];
    #pragma unroll
    for (int r = 0; r < 8; ++r) labI[r] = labels[base_i + r + hi * 8];

    float accZ[8], accS[8], accP[8];
    #pragma unroll
    for (int r = 0; r < 8; ++r) { accZ[r] = 0.f; accS[r] = 0.f; accP[r] = 0.f; }

    // --- B tile loader: 8x b128 (one clause), WMMA 16-bit B layout ---
    // lanes 0-15 : col N=lid, K = kc*32 + {0..7, 16..23}
    // lanes 16-31: col N=lid, K = kc*32 + {8..15, 24..31}
    const _Float16* bbase = fh + (size_t)(col0 + lid) * DIM + hi * 8;
    auto loadB = [&](int t) {
        BTile bt;
        const _Float16* bcol = bbase + (size_t)t * 16 * DIM;
        #pragma unroll
        for (int kc = 0; kc < 4; ++kc) {
            const v8h b0 = *(const v8h*)(bcol + kc * 32);
            const v8h b1 = *(const v8h*)(bcol + kc * 32 + 16);
            #pragma unroll
            for (int k = 0; k < 8; ++k) { bt.b[kc][k] = b0[k]; bt.b[kc][k + 8] = b1[k]; }
        }
        return bt;
    };

    BTile cur = loadB(0);
    int labj_cur = labels[col0 + lid];

    for (int t = 0; t < TILES_PER_SLICE; ++t) {
        // issue next tile's loads before consuming the current tile
        BTile nxt;
        int labj_nxt = 0;
        if (t + 1 < TILES_PER_SLICE) {
            nxt = loadB(t + 1);
            labj_nxt = labels[col0 + (t + 1) * 16 + lid];
            __builtin_prefetch(bbase + (size_t)(t + 2) * 16 * DIM, 0, 1);
        }

        v8f c = {};
        #pragma unroll
        for (int kc = 0; kc < 4; ++kc)
            c = __builtin_amdgcn_wmma_f32_16x16x32_f16(
                    false, A[kc], false, cur.b[kc], (short)0, c, false, false);

        // --- epilogue: C layout is VGPR r -> row M = r + 8*hi, lane%16 -> col
        const int j = col0 + t * 16 + lid;
        #pragma unroll
        for (int r = 0; r < 8; ++r) {
            const int   i    = base_i + r + hi * 8;
            const float x    = (c[r] - 1.0f) * INV_T;   // logits - rowmax
            const float e    = __expf(x);
            const bool  diag = (i == j);
            const bool  m    = (labI[r] == labj_cur) && !diag;
            accZ[r] += diag ? 0.0f : e;
            accS[r] += m ? x : 0.0f;
            accP[r] += m ? 1.0f : 0.0f;
        }

        cur = nxt;
        labj_cur = labj_nxt;
    }

    // reduce each row across its 16-lane half, write per-slice partials
    #pragma unroll
    for (int r = 0; r < 8; ++r) {
        #pragma unroll
        for (int msk = 8; msk >= 1; msk >>= 1) {
            accZ[r] += __shfl_xor(accZ[r], msk, 32);
            accS[r] += __shfl_xor(accS[r], msk, 32);
            accP[r] += __shfl_xor(accP[r], msk, 32);
        }
        if (lid == 0) {
            const int i = base_i + r + hi * 8;
            pZ[blockIdx.y * N_ROWS + i] = accZ[r];
            pS[blockIdx.y * N_ROWS + i] = accS[r];
            pP[blockIdx.y * N_ROWS + i] = accP[r];
        }
    }
}

// ---------------------------------------------------------------------------
// Kernel 3: combine slices (fixed order -> deterministic), per-row value,
// tree-reduce to the scalar loss.
// ---------------------------------------------------------------------------
__global__ void __launch_bounds__(256)
scl_finish(const float* __restrict__ pZ, const float* __restrict__ pS,
           const float* __restrict__ pP, float* __restrict__ out)
{
    __shared__ float red[256];
    const int t = threadIdx.x;
    float local = 0.0f;
    for (int row = t; row < N_ROWS; row += 256) {
        float Z = 0.f, S = 0.f, P = 0.f;
        #pragma unroll
        for (int s = 0; s < SLICES; ++s) {
            Z += pZ[s * N_ROWS + row];
            S += pS[s * N_ROWS + row];
            P += pP[s * N_ROWS + row];
        }
        local += S / P - logf(Z + 1e-12f);
    }
    red[t] = local;
    __syncthreads();
    #pragma unroll
    for (int off = 128; off >= 1; off >>= 1) {
        if (t < off) red[t] += red[t + off];
        __syncthreads();
    }
    if (t == 0) out[0] = -red[0] / (float)N_ROWS;
}

// ---------------------------------------------------------------------------
extern "C" void kernel_launch(void* const* d_in, const int* in_sizes, int n_in,
                              void* d_out, int out_size, void* d_ws, size_t ws_size,
                              hipStream_t stream)
{
    (void)in_sizes; (void)n_in; (void)out_size; (void)ws_size;
    const float* feats  = (const float*)d_in[0];
    const int*   labels = (const int*)d_in[1];
    float*       out    = (float*)d_out;

    _Float16* fh = (_Float16*)d_ws;                                   // 2 MB f16 features
    float* pZ = (float*)((char*)d_ws + (size_t)N_ROWS * DIM * 2);     // SLICES*N each
    float* pS = pZ + SLICES * N_ROWS;
    float* pP = pS + SLICES * N_ROWS;

    scl_normalize<<<N_ROWS / 8, 256, 0, stream>>>(feats, fh);

    dim3 g2(N_ROWS / 16 / 8, SLICES);   // 64 x 4 blocks, 8 waves each
    scl_main<<<g2, 256, 0, stream>>>(fh, labels, pZ, pS, pP);

    scl_finish<<<1, 256, 0, stream>>>(pZ, pS, pP, out);
}
